// Attention_24008867185050
// MI455X (gfx1250) — matrix operand order repaired and checked by reading, not compiled, not run on hardware
//
#include <hip/hip_runtime.h>
#include <hip/hip_bf16.h>
#include <math.h>

typedef __attribute__((ext_vector_type(16))) __bf16 v16bf;
typedef __attribute__((ext_vector_type(8)))  __bf16 v8bf;
typedef __attribute__((ext_vector_type(8)))  float  v8f;

#define BSZ  2
#define SEQ  2048
#define DIM  4096
#define NH   32
#define NKV  8
#define HD   128
#define NTOK (BSZ*SEQ)   // 4096

// ---------------------------------------------------------------------------
// Async global->LDS DMA (gfx1250 GLOBAL_LOAD_ASYNC_TO_LDS_B128, ASYNCcnt).
// ---------------------------------------------------------------------------
#if defined(__has_builtin)
#if __has_builtin(__builtin_amdgcn_global_load_async_to_lds_b128) && \
    __has_builtin(__builtin_amdgcn_s_wait_asynccnt)
#define HAS_ASYNC_LDS 1
#endif
#endif
#ifndef HAS_ASYNC_LDS
#define HAS_ASYNC_LDS 0
#endif

typedef int b128_t __attribute__((vector_size(4 * sizeof(int))));

__device__ __forceinline__ void async_copy16(const __bf16* g, __bf16* l) {
#if HAS_ASYNC_LDS
  __builtin_amdgcn_global_load_async_to_lds_b128(
      (__attribute__((address_space(1))) b128_t*)g,
      (__attribute__((address_space(3))) b128_t*)l, 0, 0);
#else
  *(v8bf*)l = *(const v8bf*)g;
#endif
}
__device__ __forceinline__ void async_wait0() {
#if HAS_ASYNC_LDS
  __builtin_amdgcn_s_wait_asynccnt(0);
#endif
}

// ---------------------------------------------------------------------------
// f32 -> bf16 elementwise conversion (grid-stride)
// ---------------------------------------------------------------------------
__global__ void cvt_f32_bf16(const float* __restrict__ in,
                             __bf16* __restrict__ out, size_t n) {
  size_t i = (size_t)blockIdx.x * blockDim.x + threadIdx.x;
  size_t stride = (size_t)gridDim.x * blockDim.x;
  for (; i < n; i += stride) out[i] = (__bf16)in[i];
}

// ---------------------------------------------------------------------------
// C[M,N] (f32) = A[M,K] (bf16 row-major) * B[N,K]^T (bf16 row-major weights)
// WG tile 128x256, 8 waves (2x4), wave tile 64x64 (16 WMMA/K-step),
// K-step 32, double-buffered LDS filled by async LDS-DMA, single barrier/step.
// ---------------------------------------------------------------------------
__global__ __launch_bounds__(256) void gemm_bf16_nt(
    const __bf16* __restrict__ A, const __bf16* __restrict__ B,
    float* __restrict__ C, int M, int N, int K)
{
  __shared__ __bf16 As[2][128 * 32];
  __shared__ __bf16 Bs[2][256 * 32];
  const int tid  = threadIdx.x;
  const int wave = tid >> 5;
  const int lane = tid & 31;
  const int hf   = lane >> 4;
  const int l16  = lane & 15;
  const int bm = blockIdx.y * 128;
  const int bn = blockIdx.x * 256;
  const int wm = (wave >> 2) * 64;  // 0 or 64
  const int wn = (wave & 3) * 64;   // 0,64,128,192

  auto issue_tile = [&](int k0, int buf) {
#pragma unroll
    for (int i = 0; i < 2; ++i) {   // A tile: 128x32 = 512 16B chunks
      int chunk = tid + i * 256;
      int row = chunk >> 2;
      int col = (chunk & 3) << 3;
      async_copy16(&A[(size_t)(bm + row) * K + k0 + col],
                   &As[buf][row * 32 + col]);
    }
#pragma unroll
    for (int i = 0; i < 4; ++i) {   // B tile: 256x32 = 1024 16B chunks
      int chunk = tid + i * 256;
      int row = chunk >> 2;
      int col = (chunk & 3) << 3;
      async_copy16(&B[(size_t)(bn + row) * K + k0 + col],
                   &Bs[buf][row * 32 + col]);
    }
  };

  v8f acc[4][4] = {};
  issue_tile(0, 0);
  int cur = 0;

  for (int k0 = 0; k0 < K; k0 += 32) {
    async_wait0();                  // own DMA for buf[cur] done
    __syncthreads();                // publishes buf[cur]; licenses buf[cur^1]
    if (k0 + 32 < K) {
      issue_tile(k0 + 32, cur ^ 1); // overlap DMA of next tile with WMMA
      __builtin_prefetch(&A[(size_t)(bm + (tid >> 1)) * K + k0 + 64], 0, 3);
      __builtin_prefetch(&B[(size_t)(bn + tid) * K + k0 + 64], 0, 3);
    }

    v16bf afrag[4];
#pragma unroll
    for (int mi = 0; mi < 4; ++mi) {
      const __bf16* p = &As[cur][(wm + mi * 16 + l16) * 32 + hf * 8];
      ((v8bf*)&afrag[mi])[0] = *(const v8bf*)p;
      ((v8bf*)&afrag[mi])[1] = *(const v8bf*)(p + 16);
    }
#pragma unroll
    for (int ni = 0; ni < 4; ++ni) {
      v16bf bfrag;
      const __bf16* p = &Bs[cur][(wn + ni * 16 + l16) * 32 + hf * 8];
      ((v8bf*)&bfrag)[0] = *(const v8bf*)p;
      ((v8bf*)&bfrag)[1] = *(const v8bf*)(p + 16);
#pragma unroll
      for (int mi = 0; mi < 4; ++mi)
        acc[mi][ni] = __builtin_amdgcn_wmma_f32_16x16x32_bf16(
            false, afrag[mi], false, bfrag, (short)0, acc[mi][ni], false, false);
    }
    cur ^= 1;
  }

#pragma unroll
  for (int mi = 0; mi < 4; ++mi)
#pragma unroll
    for (int ni = 0; ni < 4; ++ni) {
      int col  = bn + wn + ni * 16 + l16;
      int row0 = bm + wm + mi * 16 + hf * 8;
#pragma unroll
      for (int v = 0; v < 8; ++v)
        C[(size_t)(row0 + v) * N + col] = acc[mi][ni][v];
    }
}

// ---------------------------------------------------------------------------
// Fused RoPE + RMSNorm over HEAD_DIM, f32 in -> bf16 out, head-major layout.
// ---------------------------------------------------------------------------
__global__ __launch_bounds__(128) void rope_rms(
    const float* __restrict__ in,   // [NTOK][H*HD]
    const float* __restrict__ cs,   // [SEQ][HD/2]
    const float* __restrict__ sn,   // [SEQ][HD/2]
    __bf16* __restrict__ out,       // [BSZ][H][SEQ][HD]
    int H)
{
  const int wave = threadIdx.x >> 5, lane = threadIdx.x & 31;
  const int g = blockIdx.x * 4 + wave;
  const int tok = g / H, h = g % H;
  const int b = tok / SEQ, s = tok % SEQ;
  const float* row = in + (size_t)tok * H * HD + h * HD;
  __bf16* orow = out + (((size_t)b * H + h) * SEQ + s) * HD;

  const int p0 = lane, p1 = lane + 32;
  float sum = 0.f;
  float e0 = row[2 * p0], o0 = row[2 * p0 + 1];
  float c0 = cs[s * 64 + p0], s0 = sn[s * 64 + p0];
  float r0e = e0 * c0 - o0 * s0, r0o = e0 * s0 + o0 * c0;
  sum += r0e * r0e + r0o * r0o;
  float e1 = row[2 * p1], o1 = row[2 * p1 + 1];
  float c1 = cs[s * 64 + p1], s1 = sn[s * 64 + p1];
  float r1e = e1 * c1 - o1 * s1, r1o = e1 * s1 + o1 * c1;
  sum += r1e * r1e + r1o * r1o;
#pragma unroll
  for (int off = 16; off; off >>= 1) sum += __shfl_xor(sum, off, 32);
  float rs = rsqrtf(sum * (1.f / HD) + 1e-5f);
  orow[2 * p0]     = (__bf16)(r0e * rs);
  orow[2 * p0 + 1] = (__bf16)(r0o * rs);
  orow[2 * p1]     = (__bf16)(r1e * rs);
  orow[2 * p1 + 1] = (__bf16)(r1o * rs);
}

// ---------------------------------------------------------------------------
// V: f32 [NTOK][NKV*HD] -> bf16 [B][NKV][S][HD]
// ---------------------------------------------------------------------------
__global__ void v_relayout(const float* __restrict__ vf, __bf16* __restrict__ vb) {
  size_t idx = (size_t)blockIdx.x * blockDim.x + threadIdx.x;
  const size_t total = (size_t)NTOK * NKV * HD;
  for (; idx < total; idx += (size_t)gridDim.x * blockDim.x) {
    size_t tok = idx / (NKV * HD);
    int rem = (int)(idx % (NKV * HD));
    int h = rem / HD, d = rem % HD;
    int b = (int)(tok / SEQ), s = (int)(tok % SEQ);
    vb[(((size_t)b * NKV + h) * SEQ + s) * HD + d] = (__bf16)vf[idx];
  }
}

// ---------------------------------------------------------------------------
// Flash attention: one block (4 waves) per (b, head, 64-row q tile).
// Double-buffered 32-key K/V tiles (K via async LDS-DMA, V transposed in-line),
// single barrier per tile; QK^T and PV via WMMA; softmax in exp2 domain.
// ---------------------------------------------------------------------------
__global__ __launch_bounds__(128) void flash_attn(
    const __bf16* __restrict__ Q,   // [B][NH][S][HD]
    const __bf16* __restrict__ Kk,  // [B][NKV][S][HD]
    const __bf16* __restrict__ Vv,  // [B][NKV][S][HD]
    __bf16* __restrict__ O)         // [NTOK][NH*HD]
{
  __shared__ __bf16 Ks[2][32 * HD];     // [key][d]
  __shared__ __bf16 Vts[2][HD * 32];    // [d][key] (transposed)
  __shared__ __bf16 Ps[4][16 * 32];
  const int tid = threadIdx.x, wave = tid >> 5, lane = tid & 31;
  const int hf = lane >> 4, l16 = lane & 15;
  const int nqt = SEQ / 64;
  const int qt = blockIdx.x % nqt;
  const int h  = (blockIdx.x / nqt) % NH;
  const int b  = blockIdx.x / (nqt * NH);
  const int g  = h >> 2;
  const int q0 = qt * 64;
  const __bf16* qp = Q  + ((size_t)b * NH  + h) * SEQ * HD;
  const __bf16* kp = Kk + ((size_t)b * NKV + g) * SEQ * HD;
  const __bf16* vp = Vv + ((size_t)b * NKV + g) * SEQ * HD;

  auto issue_kv = [&](int kt, int buf) {
#pragma unroll
    for (int i = 0; i < 4; ++i) {
      int chunk = tid + i * 128;        // 0..511
      int row = chunk >> 4;
      int col = (chunk & 15) << 3;
      async_copy16(&kp[(size_t)(kt * 32 + row) * HD + col],
                   &Ks[buf][row * HD + col]);
      v8bf vv = *(const v8bf*)&vp[(size_t)(kt * 32 + row) * HD + col];
#pragma unroll
      for (int j = 0; j < 8; ++j) Vts[buf][(col + j) * 32 + row] = vv[j];
    }
  };

  // Q fragments for this wave's 16 rows (A layout), loaded once
  v16bf qfrag[4];
  const int qrow = q0 + wave * 16 + l16;
#pragma unroll
  for (int c = 0; c < 4; ++c) {
    const __bf16* p = qp + (size_t)qrow * HD + c * 32 + hf * 8;
    ((v8bf*)&qfrag[c])[0] = *(const v8bf*)p;
    ((v8bf*)&qfrag[c])[1] = *(const v8bf*)(p + 16);
  }

  v8f o_acc[8] = {};
  float m_s[8], l_s[8];
#pragma unroll
  for (int v = 0; v < 8; ++v) { m_s[v] = -INFINITY; l_s[v] = 0.f; }

  // score scale folded with log2(e): softmax runs in exp2 domain (native HW op)
  const float scale2 = 0.088388347648318447f * 1.4426950408889634f;
  const int nkt = (q0 + 64) / 32;
  issue_kv(0, 0);
  int cur = 0;

  for (int kt = 0; kt < nkt; ++kt) {
    async_wait0();                      // own DMA for buf[cur] done
    __syncthreads();                    // publish buf[cur]; license buf[cur^1]
    if (kt + 1 < nkt) {
      issue_kv(kt + 1, cur ^ 1);        // overlap next K/V tile with compute
      __builtin_prefetch(&kp[(size_t)((kt + 2) * 32 + (tid >> 2)) * HD], 0, 3);
    }

    // S = Q * K^T (two 16x16 score tiles over 32 keys)
    v8f s0 = {}, s1 = {};
#pragma unroll
    for (int c = 0; c < 4; ++c) {
      v16bf k0f, k1f;
      const __bf16* p0 = &Ks[cur][(l16) * HD + c * 32 + hf * 8];
      ((v8bf*)&k0f)[0] = *(const v8bf*)p0;
      ((v8bf*)&k0f)[1] = *(const v8bf*)(p0 + 16);
      const __bf16* p1 = &Ks[cur][(16 + l16) * HD + c * 32 + hf * 8];
      ((v8bf*)&k1f)[0] = *(const v8bf*)p1;
      ((v8bf*)&k1f)[1] = *(const v8bf*)(p1 + 16);
      s0 = __builtin_amdgcn_wmma_f32_16x16x32_bf16(false, qfrag[c], false, k0f,
                                                   (short)0, s0, false, false);
      s1 = __builtin_amdgcn_wmma_f32_16x16x32_bf16(false, qfrag[c], false, k1f,
                                                   (short)0, s1, false, false);
    }

    const bool need_mask = (kt * 32 + 31) > (q0 + wave * 16);
#pragma unroll
    for (int v = 0; v < 8; ++v) {
      int qr = q0 + wave * 16 + v + 8 * hf;
      float e0 = s0[v] * scale2;        // log2-domain logits
      float e1 = s1[v] * scale2;
      if (need_mask) {
        if (kt * 32 + l16 > qr)      e0 = -INFINITY;
        if (kt * 32 + 16 + l16 > qr) e1 = -INFINITY;
      }
      float rmax = fmaxf(e0, e1);
#pragma unroll
      for (int off = 1; off < 16; off <<= 1)
        rmax = fmaxf(rmax, __shfl_xor(rmax, off, 32));
      float nm = fmaxf(m_s[v], rmax);
      float alpha = __builtin_amdgcn_exp2f(m_s[v] - nm);
      float pe0 = __builtin_amdgcn_exp2f(e0 - nm);
      float pe1 = __builtin_amdgcn_exp2f(e1 - nm);
      float rsum = pe0 + pe1;
#pragma unroll
      for (int off = 1; off < 16; off <<= 1) rsum += __shfl_xor(rsum, off, 32);
      l_s[v] = l_s[v] * alpha + rsum;
      m_s[v] = nm;
#pragma unroll
      for (int t = 0; t < 8; ++t) o_acc[t][v] *= alpha;
      int prow = v + 8 * hf;
      Ps[wave][prow * 32 + l16]      = (__bf16)pe0;
      Ps[wave][prow * 32 + 16 + l16] = (__bf16)pe1;
    }

    // O += P * V  (Ps region is per-wave; same-wave LDS ops are in-order)
    v16bf pfrag;
    const __bf16* pp = &Ps[wave][l16 * 32 + hf * 8];
    ((v8bf*)&pfrag)[0] = *(const v8bf*)pp;
    ((v8bf*)&pfrag)[1] = *(const v8bf*)(pp + 16);
#pragma unroll
    for (int dt = 0; dt < 8; ++dt) {
      v16bf vfrag;
      const __bf16* vq = &Vts[cur][(dt * 16 + l16) * 32 + hf * 8];
      ((v8bf*)&vfrag)[0] = *(const v8bf*)vq;
      ((v8bf*)&vfrag)[1] = *(const v8bf*)(vq + 16);
      o_acc[dt] = __builtin_amdgcn_wmma_f32_16x16x32_bf16(
          false, pfrag, false, vfrag, (short)0, o_acc[dt], false, false);
    }
    cur ^= 1;
  }

#pragma unroll
  for (int dt = 0; dt < 8; ++dt)
#pragma unroll
    for (int v = 0; v < 8; ++v) {
      int rowloc = wave * 16 + v + 8 * hf;
      size_t tok = (size_t)b * SEQ + q0 + rowloc;
      int col = h * HD + dt * 16 + l16;
      O[tok * (NH * HD) + col] = (__bf16)(o_acc[dt][v] / l_s[v]);
    }
}

// ---------------------------------------------------------------------------
extern "C" void kernel_launch(void* const* d_in, const int* in_sizes, int n_in,
                              void* d_out, int out_size, void* d_ws, size_t ws_size,
                              hipStream_t stream) {
  const float* x    = (const float*)d_in[0];
  const float* wq   = (const float*)d_in[1];
  const float* wk   = (const float*)d_in[2];
  const float* wv   = (const float*)d_in[3];
  const float* wo   = (const float*)d_in[4];
  const float* cosp = (const float*)d_in[5];
  const float* sinp = (const float*)d_in[6];
  // d_in[7] = mask (causality implemented directly), d_in[8] = start_pos (0)

  char* ws = (char*)d_ws;
  size_t off = 0;
  auto alloc = [&](size_t bytes) {
    size_t o = off;
    off = (off + bytes + 255) & ~(size_t)255;
    return o;
  };
  __bf16* xb    = (__bf16*)(ws + alloc((size_t)NTOK * DIM * 2));
  __bf16* wqb   = (__bf16*)(ws + alloc((size_t)DIM * DIM * 2));
  __bf16* wkb   = (__bf16*)(ws + alloc((size_t)NKV * HD * DIM * 2));
  __bf16* wvb   = (__bf16*)(ws + alloc((size_t)NKV * HD * DIM * 2));
  __bf16* wob   = (__bf16*)(ws + alloc((size_t)DIM * DIM * 2));
  float*  qf    = (float*) (ws + alloc((size_t)NTOK * DIM * 4));
  float*  kf    = (float*) (ws + alloc((size_t)NTOK * NKV * HD * 4));
  float*  vf    = (float*) (ws + alloc((size_t)NTOK * NKV * HD * 4));
  __bf16* qbb   = (__bf16*)(ws + alloc((size_t)NTOK * DIM * 2));
  __bf16* kbb   = (__bf16*)(ws + alloc((size_t)NTOK * NKV * HD * 2));
  __bf16* vbb   = (__bf16*)(ws + alloc((size_t)NTOK * NKV * HD * 2));
  __bf16* attnb = (__bf16*)(ws + alloc((size_t)NTOK * DIM * 2));

  // 1) bf16 conversions
  cvt_f32_bf16<<<2048, 256, 0, stream>>>(x,  xb,  (size_t)NTOK * DIM);
  cvt_f32_bf16<<<2048, 256, 0, stream>>>(wq, wqb, (size_t)DIM * DIM);
  cvt_f32_bf16<<<2048, 256, 0, stream>>>(wk, wkb, (size_t)NKV * HD * DIM);
  cvt_f32_bf16<<<2048, 256, 0, stream>>>(wv, wvb, (size_t)NKV * HD * DIM);
  cvt_f32_bf16<<<2048, 256, 0, stream>>>(wo, wob, (size_t)DIM * DIM);

  // 2) Q/K/V projections (C = A * W^T)
  gemm_bf16_nt<<<dim3(DIM / 256, NTOK / 128), 256, 0, stream>>>(
      xb, wqb, qf, NTOK, DIM, DIM);
  gemm_bf16_nt<<<dim3((NKV * HD) / 256, NTOK / 128), 256, 0, stream>>>(
      xb, wkb, kf, NTOK, NKV * HD, DIM);
  gemm_bf16_nt<<<dim3((NKV * HD) / 256, NTOK / 128), 256, 0, stream>>>(
      xb, wvb, vf, NTOK, NKV * HD, DIM);

  // 3) RoPE + RMSNorm (Q, K) -> head-major bf16; V relayout -> bf16
  rope_rms<<<(NTOK * NH) / 4,  128, 0, stream>>>(qf, cosp, sinp, qbb, NH);
  rope_rms<<<(NTOK * NKV) / 4, 128, 0, stream>>>(kf, cosp, sinp, kbb, NKV);
  v_relayout<<<2048, 256, 0, stream>>>(vf, vbb);

  // 4) causal flash attention (WMMA QK^T and PV)
  flash_attn<<<BSZ * NH * (SEQ / 64), 128, 0, stream>>>(qbb, kbb, vbb, attnb);

  // 5) output projection -> f32 d_out
  gemm_bf16_nt<<<dim3(DIM / 256, NTOK / 128), 256, 0, stream>>>(
      attnb, wob, (float*)d_out, NTOK, DIM, DIM);
}
